// RobustPPRGo_59296318488773
// MI455X (gfx1250) — compile-verified, matrix operand order checked by reading, not run on hardware
//
#include <hip/hip_runtime.h>
#include <float.h>

typedef __attribute__((ext_vector_type(16))) __bf16 v16bf;
typedef __attribute__((ext_vector_type(8)))  __bf16 v8bf;
typedef __attribute__((ext_vector_type(8)))  float  v8f;
typedef __attribute__((ext_vector_type(2)))  float  v2f;

#if defined(__has_builtin)
#  if __has_builtin(__builtin_amdgcn_wmma_f32_16x16x4_f32)
#    define HAVE_WMMA_F32X4 1
#  endif
#endif

// ---------------------------------------------------------------- helpers ---

__device__ __forceinline__ v8f wmma_bf16(v16bf a, v16bf b, v8f c) {
  // emits v_wmma_f32_16x16x32_bf16
  return __builtin_amdgcn_wmma_f32_16x16x32_bf16(false, a, false, b, (short)0, c,
                                                 false, false);
}

// bf16 A/B fragment from bf16 memory (global or LDS), row-major, per-lane:
// lane half (lane>>4) selects K runs [kb0..kb0+7] and [kb0+16..kb0+23].
__device__ __forceinline__ v16bf load_frag_u16(const unsigned short* base, int kb0) {
  v8bf lo = *(const v8bf*)(base + kb0);
  v8bf hi = *(const v8bf*)(base + kb0 + 16);
  return __builtin_shufflevector(lo, hi, 0, 1, 2, 3, 4, 5, 6, 7,
                                 8, 9, 10, 11, 12, 13, 14, 15);
}

// bf16 A fragment converted on the fly from f32 memory.
__device__ __forceinline__ v16bf cvt_frag_f32(const float* p, int kb0) {
  float4 p0 = *(const float4*)(p + kb0);
  float4 p1 = *(const float4*)(p + kb0 + 4);
  float4 q0 = *(const float4*)(p + kb0 + 16);
  float4 q1 = *(const float4*)(p + kb0 + 20);
  v16bf a;
  a[0]  = (__bf16)p0.x; a[1]  = (__bf16)p0.y; a[2]  = (__bf16)p0.z; a[3]  = (__bf16)p0.w;
  a[4]  = (__bf16)p1.x; a[5]  = (__bf16)p1.y; a[6]  = (__bf16)p1.z; a[7]  = (__bf16)p1.w;
  a[8]  = (__bf16)q0.x; a[9]  = (__bf16)q0.y; a[10] = (__bf16)q0.z; a[11] = (__bf16)q0.w;
  a[12] = (__bf16)q1.x; a[13] = (__bf16)q1.y; a[14] = (__bf16)q1.z; a[15] = (__bf16)q1.w;
  return a;
}

__device__ __forceinline__ float wave_sum32(float x) {
#pragma unroll
  for (int m = 16; m >= 1; m >>= 1) x += __shfl_xor(x, m, 32);
  return x;
}
__device__ __forceinline__ float wave_max32(float x) {
#pragma unroll
  for (int m = 16; m >= 1; m >>= 1) x = fmaxf(x, __shfl_xor(x, m, 32));
  return x;
}

// ------------------------------------------------ weight convert/transpose ---
// dst[n*K + k] = bf16(src[k*N + n])  (store W^T so fragment loads are contiguous)
__global__ void k_cvt_transpose(const float* __restrict__ src,
                                unsigned short* __restrict__ dst, int K, int N) {
  int i = blockIdx.x * blockDim.x + threadIdx.x;
  if (i >= K * N) return;
  int n = i / K, k = i - n * K;
  __bf16 h = (__bf16)src[(size_t)k * N + n];
  dst[i] = __builtin_bit_cast(unsigned short, h);
}

// -------------------------------------------------------- fused MLP kernel ---
// logits = relu(relu(X@W1)@W2)@W3, one 16-row strip per wave, 8 waves/block.
#define H1_STRIDE 520   // 512 + 8 bf16 pad  -> bank-conflict-free frag reads
#define XCHG_STRIDE 72  // 64 + 8 bf16 pad

__global__ void __launch_bounds__(256) k_mlp(const float* __restrict__ X,
                                             const unsigned short* __restrict__ wt1,
                                             const unsigned short* __restrict__ wt2,
                                             const unsigned short* __restrict__ wt3,
                                             float* __restrict__ logits) {
  extern __shared__ char smem[];
  const int lane = threadIdx.x & 31;
  const int wave = threadIdx.x >> 5;
  const int r    = lane & 15;
  const int hb   = lane >> 4;
  const int rowBase = blockIdx.x * 128 + wave * 16;

  unsigned short* h1 = (unsigned short*)smem + wave * (16 * H1_STRIDE);
  unsigned short* xb = (unsigned short*)(smem + 8 * 16 * H1_STRIDE * 2) +
                       wave * (16 * XCHG_STRIDE);

  const v8f vzero = {0.f, 0.f, 0.f, 0.f, 0.f, 0.f, 0.f, 0.f};

  // Preload this wave's 16x256 X strip as 8 bf16 A-fragments (held in VGPRs).
  const float* xrow = X + (size_t)(rowBase + r) * 256;
  v16bf xa[8];
#pragma unroll
  for (int f = 0; f < 8; ++f) xa[f] = cvt_frag_f32(xrow, f * 32 + hb * 8);

  // ---- stage 1: H1 = relu(X @ W1)  ([16x256]@[256x512] -> LDS bf16) ----
  for (int nc = 0; nc < 8; ++nc) {
    v8f acc[4];
#pragma unroll
    for (int t = 0; t < 4; ++t) acc[t] = vzero;
#pragma unroll
    for (int f = 0; f < 8; ++f) {
      const int kb = f * 32 + hb * 8;
#pragma unroll
      for (int t = 0; t < 4; ++t) {
        const int n = nc * 64 + t * 16 + r;
        v16bf bfrag = load_frag_u16(wt1 + (size_t)n * 256, kb);
        acc[t] = wmma_bf16(xa[f], bfrag, acc[t]);
      }
    }
#pragma unroll
    for (int t = 0; t < 4; ++t)
#pragma unroll
      for (int d = 0; d < 8; ++d) {
        __bf16 h = (__bf16)fmaxf(acc[t][d], 0.0f);
        h1[(d + 8 * hb) * H1_STRIDE + nc * 64 + t * 16 + r] =
            __builtin_bit_cast(unsigned short, h);
      }
  }

  // ---- stage 2 (+ fused stage 3): H2 chunk -> relu -> logits accumulation --
  v8f lacc[4];
#pragma unroll
  for (int t = 0; t < 4; ++t) lacc[t] = vzero;

  for (int nc = 0; nc < 8; ++nc) {
    v8f acc[4];
#pragma unroll
    for (int t = 0; t < 4; ++t) acc[t] = vzero;
    for (int f = 0; f < 16; ++f) {
      const int kb = f * 32 + hb * 8;
      v16bf afrag = load_frag_u16(h1 + r * H1_STRIDE, kb);
#pragma unroll
      for (int t = 0; t < 4; ++t) {
        const int n = nc * 64 + t * 16 + r;
        v16bf bfrag = load_frag_u16(wt2 + (size_t)n * 512, kb);
        acc[t] = wmma_bf16(afrag, bfrag, acc[t]);
      }
    }
    // D-layout -> LDS (relu, bf16) so the chunk can be re-read in A-layout.
#pragma unroll
    for (int t = 0; t < 4; ++t)
#pragma unroll
      for (int d = 0; d < 8; ++d) {
        __bf16 h = (__bf16)fmaxf(acc[t][d], 0.0f);
        xb[(d + 8 * hb) * XCHG_STRIDE + t * 16 + r] =
            __builtin_bit_cast(unsigned short, h);
      }
    // stage 3 partial accumulation over K-slice [nc*64, nc*64+64)
#pragma unroll
    for (int f3 = 0; f3 < 2; ++f3) {
      const int kbl = f3 * 32 + hb * 8;
      v16bf a3 = load_frag_u16(xb + r * XCHG_STRIDE, kbl);
#pragma unroll
      for (int t = 0; t < 4; ++t) {
        const int n = t * 16 + r;
        v16bf b3 = load_frag_u16(wt3 + (size_t)n * 512, nc * 64 + kbl);
        lacc[t] = wmma_bf16(a3, b3, lacc[t]);
      }
    }
  }

#pragma unroll
  for (int t = 0; t < 4; ++t)
#pragma unroll
    for (int d = 0; d < 8; ++d)
      logits[(size_t)(rowBase + d + 8 * hb) * 64 + t * 16 + r] = lacc[t][d];
}

// ----------------------------------------------------- soft-k-medoid kernel --
// One wave per output row b. K=32 selected of 64, c=64, TEMPERATURE=1.
#define XK_STRIDE 68  // 64 + 4 f32 pad (16B aligned, bank-conflict-free)
#define MED_WSLOT 2400

__global__ void __launch_bounds__(256) k_medoid(const float* __restrict__ logits,
                                                const int* __restrict__ ppr_idx,
                                                const float* __restrict__ ppr_vals,
                                                float* __restrict__ out) {
  extern __shared__ char smem[];
  const int lane = threadIdx.x & 31;
  const int wave = threadIdx.x >> 5;
  const int b    = blockIdx.x * 8 + wave;
  const int r    = lane & 15;
  const int hb   = lane >> 4;

  float* base   = (float*)smem + wave * MED_WSLOT;
  float* vals64 = base;                 // 64
  float* wkb    = base + 64;            // 32 selected weights
  int*   ikb    = (int*)(base + 96);    // 32 selected indices
  float* sqb    = base + 128;           // 32 squared norms
  float* distb  = base + 160;           // 32 weighted distances (atomic)
  float* smb    = base + 192;           // 32 softmax weights
  float* xk     = base + 224;           // 32 x XK_STRIDE gathered logits

  // 1) load PPR weights, rank for stable-descending top-32 selection
  const float v0 = ppr_vals[(size_t)b * 64 + lane];
  const float v1 = ppr_vals[(size_t)b * 64 + 32 + lane];
  vals64[lane] = v0;
  vals64[lane + 32] = v1;
  distb[lane] = 0.0f;
  __syncthreads();
  int c0 = 0, c1 = 0;
  for (int t = 0; t < 64; ++t) {
    const float u = vals64[t];
    c0 += (u > v0) || (u == v0 && t < lane);
    c1 += (u > v1) || (u == v1 && t < lane + 32);
  }
  if (c0 < 32) { wkb[c0] = v0; ikb[c0] = ppr_idx[(size_t)b * 64 + lane]; }
  if (c1 < 32) { wkb[c1] = v1; ikb[c1] = ppr_idx[(size_t)b * 64 + 32 + lane]; }
  __syncthreads();

  // 2) gather neighbor logits (lane j owns row j), squared norms
  const float wj = wkb[lane];
  const float* lr = logits + (size_t)ikb[lane] * 64;
  float ssq = 0.f;
#pragma unroll
  for (int c = 0; c < 64; c += 4) {
    float4 p = *(const float4*)(lr + c);
    ssq += p.x * p.x + p.y * p.y + p.z * p.z + p.w * p.w;
    *(float4*)(&xk[lane * XK_STRIDE + c]) = p;
  }
  sqb[lane] = ssq;
  __syncthreads();

  // 3) Gram g = xk @ xk^T via exact-f32 WMMA; fold into weighted L2 distances
  for (int tj = 0; tj < 2; ++tj)
    for (int tm = 0; tm < 2; ++tm) {
      v8f g = {0.f, 0.f, 0.f, 0.f, 0.f, 0.f, 0.f, 0.f};
#if defined(HAVE_WMMA_F32X4)
#pragma unroll
      for (int k = 0; k < 64; k += 4) {
        const int kb = k + hb * 2;
        v2f a, bb;
        a[0]  = xk[(tj * 16 + r) * XK_STRIDE + kb];
        a[1]  = xk[(tj * 16 + r) * XK_STRIDE + kb + 1];
        bb[0] = xk[(tm * 16 + r) * XK_STRIDE + kb];
        bb[1] = xk[(tm * 16 + r) * XK_STRIDE + kb + 1];
        g = __builtin_amdgcn_wmma_f32_16x16x4_f32(false, a, false, bb, (short)0, g,
                                                  false, false);
      }
#else
#pragma unroll
      for (int d = 0; d < 8; ++d) {
        const int j = tj * 16 + d + 8 * hb, m = tm * 16 + r;
        float s = 0.f;
        for (int c = 0; c < 64; ++c) s += xk[j * XK_STRIDE + c] * xk[m * XK_STRIDE + c];
        g[d] = s;
      }
#endif
      const float sqm = sqb[tm * 16 + r];
      const float wm  = wkb[tm * 16 + r];
#pragma unroll
      for (int d = 0; d < 8; ++d) {
        const int j = tj * 16 + d + 8 * hb;
        const float l2 = sqrtf(fmaxf(sqb[j] + sqm - 2.0f * g[d], 0.0f));
        atomicAdd(&distb[j], l2 * wm);
      }
    }
  __syncthreads();

  // 4) masked softmax with weight correction (lane j owns entry j)
  float dj = distb[lane];
  if (wj == 0.f) dj = FLT_MAX;
  const float rs = wave_sum32(wj);
  if (rs == 0.f) dj = 0.f;
  const float s  = -dj / rs;  // TEMPERATURE == 1
  const float mx = wave_max32(s);
  const float e  = __expf(s - mx);
  const float Z  = wave_sum32(e);
  float smv = (e / Z) * wj;
  const float Z2 = wave_sum32(smv);
  smv /= Z2;
  smb[lane] = smv;
  __syncthreads();

  // 5) out[b] = rs * sum_j sm[j] * xk[j,:]; lane handles two columns
  const int cc = lane * 2;
  float a0 = 0.f, a1 = 0.f;
#pragma unroll
  for (int j = 0; j < 32; ++j) {
    const float sv = smb[j];
    a0 += sv * xk[j * XK_STRIDE + cc];
    a1 += sv * xk[j * XK_STRIDE + cc + 1];
  }
  out[(size_t)b * 64 + cc]     = rs * a0;
  out[(size_t)b * 64 + cc + 1] = rs * a1;
}

// ------------------------------------------------------------------- launch --
extern "C" void kernel_launch(void* const* d_in, const int* in_sizes, int n_in,
                              void* d_out, int out_size, void* d_ws, size_t ws_size,
                              hipStream_t stream) {
  (void)in_sizes; (void)n_in; (void)out_size; (void)ws_size;
  const float* X        = (const float*)d_in[0];
  const int*   ppr_idx  = (const int*)d_in[1];
  const float* ppr_vals = (const float*)d_in[2];
  const float* W1       = (const float*)d_in[3];
  const float* W2       = (const float*)d_in[4];
  const float* W3       = (const float*)d_in[5];

  char* ws = (char*)d_ws;
  unsigned short* wt1 = (unsigned short*)ws;                              // [512][256] bf16
  unsigned short* wt2 = (unsigned short*)(ws + 512 * 256 * 2);            // [512][512] bf16
  unsigned short* wt3 = (unsigned short*)(ws + 512 * 256 * 2 + 512 * 512 * 2); // [64][512]
  float* logits = (float*)(ws + (1 << 20));                               // [131072][64] f32

  k_cvt_transpose<<<(512 * 256 + 255) / 256, 256, 0, stream>>>(W1, wt1, 256, 512);
  k_cvt_transpose<<<(512 * 512 + 255) / 256, 256, 0, stream>>>(W2, wt2, 512, 512);
  k_cvt_transpose<<<(512 * 64 + 255) / 256, 256, 0, stream>>>(W3, wt3, 512, 64);

  const size_t mlp_lds = 8 * 16 * H1_STRIDE * 2 + 8 * 16 * XCHG_STRIDE * 2; // 151552 B
  k_mlp<<<131072 / 128, 256, mlp_lds, stream>>>(X, wt1, wt2, wt3, logits);

  const size_t med_lds = 8 * MED_WSLOT * sizeof(float); // 76800 B
  k_medoid<<<32768 / 8, 256, med_lds, stream>>>(logits, ppr_idx, ppr_vals,
                                                (float*)d_out);
}
